// Conv2d_78331613544815
// MI455X (gfx1250) — matrix operand order repaired and checked by reading, not compiled, not run on hardware
//
#include <hip/hip_runtime.h>

// Implicit-GEMM 3x3 conv for MI455X (gfx1250, wave32, WMMA + TDM).
// M = Cout = 256, N = B*H*W = 100352, K = Cin*9 = 1152.
// bf16 operands (fp32->bf16 RNE), fp32 accumulation via v_wmma_f32_16x16x32_bf16.
// ~380 FLOP/byte => compute-bound. Organization:
//  - 32x32 C-tile per wave: 4 WMMAs per k-step, every fragment reused twice
//  - K-chunks of 64 (two k-steps per LDS tile) => 18 iterations
//  - A (weights) tile DMA'd by the Tensor Data Mover into LDS with HW row
//    padding (pad_interval=32dw, pad_amount=8dw -> 160B stride), overlapped
//    with compute; completion via s_wait_tensorcnt + workgroup barrier
//  - B (im2col) gathered on the fly, register-staged double buffering

typedef __attribute__((ext_vector_type(16))) __bf16 v16bf;
typedef __attribute__((ext_vector_type(8)))  __bf16 v8bf;
typedef __attribute__((ext_vector_type(8)))  float  v8f;
typedef __attribute__((ext_vector_type(4)))  unsigned u32x4;
typedef __attribute__((ext_vector_type(8)))  int      i32x8;
typedef __attribute__((ext_vector_type(4)))  int      i32x4;

#define CIN    128
#define COUT   256
#define HW     3136
#define IMW    56
#define LDK    80     // LDS row stride in bf16: 64 data + 16 pad = 160B (16B aligned)
#define NCHUNK 18     // 9 taps * 2 ci-chunks of 64

#if __has_builtin(__builtin_amdgcn_tensor_load_to_lds) && __has_builtin(__builtin_amdgcn_s_wait_tensorcnt)
#define HAVE_TDM 1
#else
#define HAVE_TDM 0
#endif

__device__ __forceinline__ __bf16 f2bf(float f) {
    unsigned u = __builtin_bit_cast(unsigned, f);
    u += 0x7FFFu + ((u >> 16) & 1u);
    unsigned short h = (unsigned short)(u >> 16);
    return __builtin_bit_cast(__bf16, h);
}

// ---- one-time weight repack: OIHW fp32 -> [tap][cic][co][k64] bf16 -------
// w3 flat index = ((tap*2 + cic)*256 + co)*64 + k   (ci = cic*64 + k)
__global__ void repack_w_kernel(const float* __restrict__ k_oihw,
                                __bf16* __restrict__ w3) {
    int i  = blockIdx.x * 256 + threadIdx.x;  // 294912 elements exactly
    int k  = i & 63;
    int t1 = i >> 6;
    int co = t1 & 255;
    int t2 = t1 >> 8;        // tap*2 + cic
    int ci = (t2 & 1) * 64 + k;
    int tap = t2 >> 1;
    w3[i] = f2bf(k_oihw[(co * CIN + ci) * 9 + tap]);
}

// ---- main conv kernel ----------------------------------------------------
__global__ void __launch_bounds__(256)
conv_wmma_kernel(const float* __restrict__ x,       // [32][128][56][56]
                 const __bf16* __restrict__ w3,     // [9][2][256][64] bf16
                 const float* __restrict__ bias,    // [256]
                 float* __restrict__ out) {         // [32][256][56][56]
    __shared__ __bf16 Asm[2][128 * LDK];  // weights tile: [co within 128][k 0..63]
    __shared__ __bf16 Bsm[2][64 * LDK];   // im2col tile:  [pixel within 64][k 0..63]

    const int tid    = threadIdx.x;
    const int lane   = tid & 31;
    const int wid    = tid >> 5;       // wave 0..7
    const int laneLo = lane & 15;
    const int laneHi = lane >> 4;

    const int pixBase = blockIdx.x * 64;    // 49 tiles of 64 pixels (3136 = 49*64)
    const int coBase  = blockIdx.y * 128;   // 2 tiles of 128 Cout
    const int n       = blockIdx.z;         // batch

    // B staging: each thread owns one pixel, 16 k values (two v8bf)
    const int pLoc = tid & 63;
    const int kb   = (tid >> 6) * 16;
    const int pix  = pixBase + pLoc;
    const int h    = pix / IMW;
    const int w    = pix % IMW;

    const float* xn = x + (long)n * CIN * HW;

    // per-wave output sub-tile: 32 co rows x 32 pixel cols (4x2 wave grid)
    const int mWave = (wid & 3) * 32;
    const int nWave = (wid >> 2) * 32;

    v8f c00 = {}, c01 = {}, c10 = {}, c11 = {};

    v8bf bRegV[2];
#if !HAVE_TDM
    const int aCo = tid >> 1;           // fallback A staging
    const int akb = (tid & 1) * 32;
    v8bf aRegV[4];
#endif

    auto stageB = [&](int chunk) {
        const int tap = chunk >> 1;          // 0..8
        const int cic = chunk & 1;           // ci block of 64
        const int ih  = h + (tap / 3) - 1;
        const int iw  = w + (tap % 3) - 1;
        const bool valid = ((unsigned)ih < 56u) && ((unsigned)iw < 56u);
        const long xoff = (long)(cic * 64 + kb) * HW + (long)ih * IMW + iw;
#pragma unroll
        for (int j = 0; j < 16; ++j)
            bRegV[j >> 3][j & 7] = f2bf(valid ? xn[xoff + (long)j * HW] : 0.0f);
#if !HAVE_TDM
        const v8bf* wp = (const v8bf*)(w3 + (((long)(tap * 2 + cic) * COUT
                                              + coBase + aCo) << 6) + akb);
#pragma unroll
        for (int j = 0; j < 4; ++j) aRegV[j] = wp[j];
#endif
    };

    auto commitB = [&](int buf) {
        *(v8bf*)&Bsm[buf][pLoc * LDK + kb]     = bRegV[0];
        *(v8bf*)&Bsm[buf][pLoc * LDK + kb + 8] = bRegV[1];
#if !HAVE_TDM
#pragma unroll
        for (int j = 0; j < 4; ++j)
            *(v8bf*)&Asm[buf][aCo * LDK + akb + 8 * j] = aRegV[j];
#endif
    };

#if HAVE_TDM
    // TDM: DMA one 128(co) x 64(k) bf16 tile from prepacked weights into LDS,
    // inserting 32B of padding after every 128B row (-> LDK=80 row stride).
    auto tdmLoadA = [&](int chunk, int buf) {
        const int tap = chunk >> 1, cic = chunk & 1;
        unsigned long ga = (unsigned long)(w3 + (((long)(tap * 2 + cic) * COUT + coBase) << 6));
        unsigned lds = (unsigned)(unsigned long)&Asm[buf][0];  // LDS aperture: low 32 bits = LDS offset
        u32x4 g0;
        g0[0] = 1u;                                   // count=1 (valid user descriptor)
        g0[1] = lds;                                  // lds_addr (bytes)
        g0[2] = (unsigned)ga;                         // global_addr[31:0]
        g0[3] = (unsigned)(ga >> 32) | (2u << 30);    // global_addr[56:32] | type=2 ("image")
        i32x8 g1;
        g1[0] = (int)((1u << 16)      // data_size = 1 -> 2 bytes
                    | (1u << 20)      // pad_enable
                    | (4u << 22)      // pad_interval: 32 DWORDs (=128B = one 64-elem row)
                    | (7u << 25));    // pad_amount:   8 DWORDs (=32B)
        g1[1] = (int)(64u << 16);     // tensor_dim0 = 64  (bits 79:48, low half)
        g1[2] = (int)(128u << 16);    // tensor_dim0 hi = 0; tensor_dim1 = 128 (low half)
        g1[3] = (int)(64u << 16);     // tensor_dim1 hi = 0; tile_dim0 = 64
        g1[4] = 128;                  // tile_dim1 = 128; tile_dim2 = 0
        g1[5] = 64;                   // tensor_dim0_stride = 64 (low 32)
        g1[6] = 0;                    // stride hi; tensor_dim1_stride lo
        g1[7] = 0;
        i32x4 z4 = {0, 0, 0, 0};      // 2-D tensor: groups 2/3 unused
#if defined(__clang_major__) && __clang_major__ >= 23
        i32x8 z8 = {0, 0, 0, 0, 0, 0, 0, 0};
        __builtin_amdgcn_tensor_load_to_lds(g0, g1, z4, z4, z8, 0);
#else
        __builtin_amdgcn_tensor_load_to_lds(g0, g1, z4, z4, 0);
#endif
    };
#endif

    auto load_a = [&](const __bf16* base, int row, int koff) -> v16bf {
        // A 16x32 (contiguous k, matching B): lanes 0-15 -> K 0-15, lanes 16-31 -> K 16-31
        const __bf16* ap = base + row * LDK + koff + laneHi * 16;
        return *(const v16bf*)ap;
    };

    // ---- prologue: tile 0 ----
#if HAVE_TDM
    if (wid == 0) tdmLoadA(0, 0);
#endif
    stageB(0);
    commitB(0);
#if HAVE_TDM
    if (wid == 0) __builtin_amdgcn_s_wait_tensorcnt(0);
#endif
    __syncthreads();

    for (int ch = 0; ch < NCHUNK; ++ch) {
        const int buf  = ch & 1;
        const int nbuf = buf ^ 1;
        if (ch + 1 < NCHUNK) {
#if HAVE_TDM
            if (wid == 0) tdmLoadA(ch + 1, nbuf);  // DMA next A tile while we compute
#endif
            stageB(ch + 1);                        // prefetch next B chunk into registers
        }

#pragma unroll
        for (int s = 0; s < 2; ++s) {              // two WMMA k-steps per LDS tile
            const int ko = s * 32;
            v16bf a0 = load_a(Asm[buf], mWave + laneLo, ko);
            v16bf a1 = load_a(Asm[buf], mWave + 16 + laneLo, ko);
            // B KxN in contiguous k order (same as A): lanes 0-15 -> K 0-15, lanes 16-31 -> K 16-31:
            // one contiguous 32B LDS load per lane (tile stored N-major).
            v16bf b0 = *(const v16bf*)&Bsm[buf][(nWave + laneLo) * LDK + ko + laneHi * 16];
            v16bf b1 = *(const v16bf*)&Bsm[buf][(nWave + 16 + laneLo) * LDK + ko + laneHi * 16];

            c00 = __builtin_amdgcn_wmma_f32_16x16x32_bf16(false, a0, false, b0, (short)0, c00, false, false);
            c01 = __builtin_amdgcn_wmma_f32_16x16x32_bf16(false, a0, false, b1, (short)0, c01, false, false);
            c10 = __builtin_amdgcn_wmma_f32_16x16x32_bf16(false, a1, false, b0, (short)0, c10, false, false);
            c11 = __builtin_amdgcn_wmma_f32_16x16x32_bf16(false, a1, false, b1, (short)0, c11, false, false);
        }

        __syncthreads();                        // all reads of buf done
        if (ch + 1 < NCHUNK) commitB(nbuf);     // publish next B tile
#if HAVE_TDM
        if (wid == 0 && ch + 1 < NCHUNK)
            __builtin_amdgcn_s_wait_tensorcnt(0);  // next A tile landed in LDS
#endif
        __syncthreads();                        // publish both tiles to all waves
    }

    // Epilogue. C/D layout: VGPR r, lanes 0-15 -> (M=r, N=lane); lanes 16-31 -> (M=8+r, N=lane-16).
#pragma unroll
    for (int r = 0; r < 8; ++r) {
        const int M  = r + (laneHi << 3);
        const int p0 = pixBase + nWave + laneLo;

        {   // M-tile 0 (rows mWave .. mWave+15)
            const int co = coBase + mWave + M;
            const float bv = bias[co];
            const long o = ((long)n * COUT + co) * HW;
            out[o + p0]      = c00[r] + bv;
            out[o + p0 + 16] = c01[r] + bv;
        }
        {   // M-tile 1 (rows mWave+16 .. mWave+31)
            const int co = coBase + mWave + 16 + M;
            const float bv = bias[co];
            const long o = ((long)n * COUT + co) * HW;
            out[o + p0]      = c10[r] + bv;
            out[o + p0 + 16] = c11[r] + bv;
        }
    }
}

extern "C" void kernel_launch(void* const* d_in, const int* in_sizes, int n_in,
                              void* d_out, int out_size, void* d_ws, size_t ws_size,
                              hipStream_t stream) {
    (void)in_sizes; (void)n_in; (void)out_size; (void)ws_size;
    const float* x    = (const float*)d_in[0];
    const float* kern = (const float*)d_in[1];
    const float* bias = (const float*)d_in[2];
    float* out        = (float*)d_out;
    __bf16* w3        = (__bf16*)d_ws;   // 9*2*256*64*2 = 576 KB of scratch

    // 1) repack weights into the TDM/LDS tile layout (294912 / 256 = 1152 blocks)
    repack_w_kernel<<<dim3(1152), dim3(256), 0, stream>>>(kern, w3);
    // 2) conv: grid = (pixel tiles, cout tiles, batch)
    conv_wmma_kernel<<<dim3(49, 2, 32), dim3(256), 0, stream>>>(x, w3, bias, out);
}
